// GAT_70987219468862
// MI455X (gfx1250) — matrix operand order repaired and checked by reading, not compiled, not run on hardware
//
#include <hip/hip_runtime.h>
#include <hip/hip_bf16.h>
#include <math.h>

typedef __attribute__((ext_vector_type(16))) __bf16 v16bf;
typedef __attribute__((ext_vector_type(8)))  float  v8f;

#define ALPHA 0.2f
#define MASKV -9e15f

#if defined(__HIP_DEVICE_COMPILE__) && __has_builtin(__builtin_amdgcn_global_load_async_to_lds_b128)
#define HAVE_ASYNC_LDS 1
#else
#define HAVE_ASYNC_LDS 0
#endif

union FragBF { v16bf v; __bf16 h[16]; uint4 q[2]; };

__device__ __forceinline__ float lrelu(float t) { return t > 0.f ? t : ALPHA * t; }

#if HAVE_ASYNC_LDS
typedef int ai4 __attribute__((vector_size(16)));
__device__ __forceinline__ void async_copy_b128(const void* g, void* l) {
  __builtin_amdgcn_global_load_async_to_lds_b128(
      (__attribute__((address_space(1))) ai4*)g,
      (__attribute__((address_space(3))) ai4*)l, 0, 0);
}
#endif

// ---------------------------------------------------------------- converts
__global__ __launch_bounds__(256) void k_cvt_x(const float* __restrict__ x,
                                               __bf16* __restrict__ xb, int n4) {
  int t = blockIdx.x * 256 + threadIdx.x;
  if (t >= n4) return;
  float4 v = ((const float4*)x)[t];
  union { uint2 u; __bf16 h[4]; } o;
  o.h[0] = (__bf16)v.x; o.h[1] = (__bf16)v.y; o.h[2] = (__bf16)v.z; o.h[3] = (__bf16)v.w;
  ((uint2*)xb)[t] = o.u;
}

__global__ __launch_bounds__(256) void k_cvt_wh(const float* __restrict__ W,
                                                const float* __restrict__ HW,
                                                __bf16* __restrict__ Wb,
                                                __bf16* __restrict__ HWb) {
  int t = blockIdx.x * 256 + threadIdx.x;   // t < 131072
  if (t < 65536) Wb[t] = (__bf16)W[t];
  else           HWb[t - 65536] = (__bf16)HW[t - 65536];
}

// ------------------------------------------------------------- a1/a2 GEMVs
__global__ __launch_bounds__(256) void k_rowdots(const float* __restrict__ x,
                                                 const float* __restrict__ a,
                                                 float* __restrict__ a1,
                                                 float* __restrict__ a2, int D) {
  size_t r = (size_t)blockIdx.x * 256 + threadIdx.x;
  const float* xr = x + r * D;
  float s1 = 0.f, s2 = 0.f;
  for (int k = 0; k < D; k += 4) {
    float4 xv = *(const float4*)(xr + k);
    float4 v1 = *(const float4*)(a + k);
    float4 v2 = *(const float4*)(a + D + k);
    s1 += xv.x * v1.x + xv.y * v1.y + xv.z * v1.z + xv.w * v1.w;
    s2 += xv.x * v2.x + xv.y * v2.y + xv.z * v2.z + xv.w * v2.w;
  }
  a1[r] = s1; a2[r] = s2;
}

// ---------------------------------------------- column softmax stats (axis=1)
__global__ __launch_bounds__(256) void k_stats_part(const int* __restrict__ adj,
                                                    const float* __restrict__ a1,
                                                    const float* __restrict__ a2,
                                                    float* __restrict__ mpart,
                                                    float* __restrict__ spart, int Nn) {
  const int b = blockIdx.z, ic = blockIdx.y, jt = blockIdx.x, tid = threadIdx.x;
  const int j = jt * 256 + tid;
  __shared__ float A1S[256];
  A1S[tid] = a1[(size_t)b * Nn + ic * 256 + tid];
  __syncthreads();
  const float a2j = a2[(size_t)b * Nn + j];
  const int* col = adj + ((size_t)b * Nn + (size_t)ic * 256) * Nn + j;
  float m = -INFINITY, s = 0.f;
  for (int ii = 0; ii < 256; ++ii) {
    int av = col[(size_t)ii * Nn];
    float t = A1S[ii] + a2j;
    float v = (av > 0) ? lrelu(t) : MASKV;
    float mn = fmaxf(m, v);
    s = s * __expf(m - mn) + __expf(v - mn);
    m = mn;
  }
  mpart[((size_t)b * 8 + ic) * Nn + j] = m;
  spart[((size_t)b * 8 + ic) * Nn + j] = s;
}

__global__ __launch_bounds__(256) void k_stats_comb(const float* __restrict__ mpart,
                                                    const float* __restrict__ spart,
                                                    float* __restrict__ mfin,
                                                    float* __restrict__ sinv, int Nn) {
  size_t t = (size_t)blockIdx.x * 256 + threadIdx.x;   // t < B*N
  size_t b = t / Nn, j = t - b * Nn;
  float m = -INFINITY, s = 0.f;
  for (int c = 0; c < 8; ++c) {
    float mm = mpart[(b * 8 + c) * Nn + j];
    float ss = spart[(b * 8 + c) * Nn + j];
    float mn = fmaxf(m, mm);
    s = s * __expf(m - mn) + ss * __expf(mm - mn);
    m = mn;
  }
  mfin[t] = m;
  sinv[t] = 1.f / s;
}

// --------------------------------- WMMA GEMM: C[M x 256] = A[M x K] * B[K x 256]
// MODE 0: store bf16 to outB.   MODE 1: store f32 + bias to outF.
template <int MODE>
__global__ __launch_bounds__(256) void k_gemm256(const __bf16* __restrict__ A,
                                                 const __bf16* __restrict__ Bm,
                                                 __bf16* __restrict__ outB,
                                                 float* __restrict__ outF,
                                                 const float* __restrict__ bias,
                                                 int M, int K) {
  const int b = blockIdx.y, i0 = blockIdx.x * 64, tid = threadIdx.x;
  const int lane = tid & 31, wid = tid >> 5;
  const int isub = (wid >> 1) * 16, dhalf = (wid & 1) * 128;
  const int mrow = lane & 15, khalf = lane >> 4;
  const int kb = khalf * 8, kr = khalf * 16;

  __shared__ __bf16 As[2][64 * 40];
  __shared__ __bf16 BT[256 * 40];

  const __bf16* Ab = A + (size_t)b * M * K;
  const int ar = tid >> 2, ap = (tid & 3) * 8;    // A staging coords
  const int bk = tid >> 3, bn = (tid & 7) * 32;   // B staging coords

  v8f acc[8];
  v8f z = {0.f, 0.f, 0.f, 0.f, 0.f, 0.f, 0.f, 0.f};
#pragma unroll
  for (int t = 0; t < 8; ++t) acc[t] = z;

  const int nsteps = K / 32;
  union BReg { uint4 q[4]; __bf16 h[32]; } breg, bnext;

#if HAVE_ASYNC_LDS
  async_copy_b128(Ab + (size_t)(i0 + ar) * K + ap, &As[0][ar * 40 + ap]);
#else
  uint4 areg = *(const uint4*)(Ab + (size_t)(i0 + ar) * K + ap);
  uint4 aregn = areg;
#endif
  {
    const uint4* src = (const uint4*)(Bm + (size_t)bk * 256 + bn);
    breg.q[0] = src[0]; breg.q[1] = src[1]; breg.q[2] = src[2]; breg.q[3] = src[3];
  }
  bnext = breg;

  for (int s = 0; s < nsteps; ++s) {
    const int cur = s & 1;
#if !HAVE_ASYNC_LDS
    *(uint4*)(&As[cur][ar * 40 + ap]) = areg;
#endif
#pragma unroll
    for (int q = 0; q < 32; ++q) BT[(bn + q) * 40 + bk] = breg.h[q];
    if (s + 1 < nsteps) {   // prefetch next tile while this one computes
      const int k0n = (s + 1) * 32;
#if HAVE_ASYNC_LDS
      async_copy_b128(Ab + (size_t)(i0 + ar) * K + k0n + ap, &As[1 - cur][ar * 40 + ap]);
#else
      aregn = *(const uint4*)(Ab + (size_t)(i0 + ar) * K + k0n + ap);
#endif
      const uint4* src = (const uint4*)(Bm + (size_t)(k0n + bk) * 256 + bn);
      bnext.q[0] = src[0]; bnext.q[1] = src[1]; bnext.q[2] = src[2]; bnext.q[3] = src[3];
    }
#if HAVE_ASYNC_LDS
    if (s + 1 < nsteps) asm volatile("s_wait_asynccnt 0x1" ::: "memory");
    else                asm volatile("s_wait_asynccnt 0x0" ::: "memory");
#endif
    __syncthreads();
    FragBF af;
    af.q[0] = *(const uint4*)(&As[cur][(isub + mrow) * 40 + kb]);
    af.q[1] = *(const uint4*)(&As[cur][(isub + mrow) * 40 + kb + 16]);
#pragma unroll
    for (int dt = 0; dt < 8; ++dt) {
      FragBF bfm;
      const __bf16* bp = &BT[(dhalf + dt * 16 + mrow) * 40 + kb];
      bfm.q[0] = *(const uint4*)(bp);
      bfm.q[1] = *(const uint4*)(bp + 16);
      acc[dt] = __builtin_amdgcn_wmma_f32_16x16x32_bf16(
          false, af.v, false, bfm.v, (short)0, acc[dt], false, false);
    }
    breg = bnext;
#if !HAVE_ASYNC_LDS
    areg = aregn;
#endif
    __syncthreads();
  }

  const size_t oBase = (size_t)b * M * 256 + (size_t)(i0 + isub + khalf * 8) * 256;
#pragma unroll
  for (int dt = 0; dt < 8; ++dt) {
    const int dcol = dhalf + dt * 16 + mrow;
    if (MODE == 1) {
      const float bv = bias[dcol];
      float* p = outF + oBase + dcol;
#pragma unroll
      for (int r = 0; r < 8; ++r) p[(size_t)r * 256] = acc[dt][r] + bv;
    } else {
      __bf16* p = outB + oBase + dcol;
#pragma unroll
      for (int r = 0; r < 8; ++r) p[(size_t)r * 256] = (__bf16)acc[dt][r];
    }
  }
}

// ---------------- fused attention GEMM: hid = ELU( softmax_col(mask(e)) @ h )
__global__ __launch_bounds__(256) void k_attn(const int* __restrict__ adj,
                                              const __bf16* __restrict__ hB,
                                              const float* __restrict__ a1,
                                              const float* __restrict__ a2,
                                              const float* __restrict__ mfin,
                                              const float* __restrict__ sinv,
                                              __bf16* __restrict__ hid, int Nn) {
  const int b = blockIdx.y, i0 = blockIdx.x * 64, tid = threadIdx.x;
  const int lane = tid & 31, wid = tid >> 5;
  const int isub = (wid >> 1) * 16, dhalf = (wid & 1) * 128;
  const int mrow = lane & 15, khalf = lane >> 4;
  const int kb = khalf * 8, kr = khalf * 16;

  __shared__ __bf16 BT[256 * 40];
  __shared__ alignas(16) unsigned char AM[64 * 32];
  __shared__ float4 JD[32];   // per-j: {a2, P=e^(a2-m)/s, Q=e^(0.2a2-m)/s, -}

  const size_t bN = (size_t)b * Nn;
  const float a1v = a1[bN + i0 + isub + mrow];
  const float E1 = __expf(a1v), E2 = __expf(ALPHA * a1v);

  const int hj = tid >> 3, hd = (tid & 7) * 32;   // h-panel staging coords
  const int al = tid >> 2, ap = (tid & 3) * 8;    // adj staging coords
  const int* adjRow = adj + (bN + i0 + al) * Nn;

  v8f acc[8];
  v8f z = {0.f, 0.f, 0.f, 0.f, 0.f, 0.f, 0.f, 0.f};
#pragma unroll
  for (int t = 0; t < 8; ++t) acc[t] = z;

  union HReg { uint4 q[4]; __bf16 h[32]; } hreg, hnext;
  int4 m0, m1, m0n, m1n;
  float sa2 = 0.f, sP = 0.f, sQ = 0.f, sa2n = 0.f, sPn = 0.f, sQn = 0.f;

  { // prologue: tile 0 into registers
    const uint4* src = (const uint4*)(hB + (bN + hj) * 256 + hd);
    hreg.q[0] = src[0]; hreg.q[1] = src[1]; hreg.q[2] = src[2]; hreg.q[3] = src[3];
    m0 = *(const int4*)(adjRow + ap);
    m1 = *(const int4*)(adjRow + ap + 4);
    if (tid < 32) {
      float a2v = a2[bN + tid], mv = mfin[bN + tid], sv = sinv[bN + tid];
      sa2 = a2v; sP = __expf(a2v - mv) * sv; sQ = __expf(ALPHA * a2v - mv) * sv;
    }
  }
  hnext = hreg; m0n = m0; m1n = m1;

  const int nst = Nn / 32;
  for (int s = 0; s < nst; ++s) {
    // commit staged registers to LDS
#pragma unroll
    for (int q = 0; q < 32; ++q) BT[(hd + q) * 40 + hj] = hreg.h[q];
    {
      unsigned lo = (m0.x > 0 ? 1u : 0u) | (m0.y > 0 ? 0x100u : 0u) |
                    (m0.z > 0 ? 0x10000u : 0u) | (m0.w > 0 ? 0x1000000u : 0u);
      unsigned hi = (m1.x > 0 ? 1u : 0u) | (m1.y > 0 ? 0x100u : 0u) |
                    (m1.z > 0 ? 0x10000u : 0u) | (m1.w > 0 ? 0x1000000u : 0u);
      *(uint2*)(AM + al * 32 + ap) = make_uint2(lo, hi);
    }
    if (tid < 32) JD[tid] = make_float4(sa2, sP, sQ, 0.f);
    if (s + 1 < nst) {   // prefetch next tile into registers
      const int j1 = (s + 1) * 32;
      const uint4* src = (const uint4*)(hB + (bN + j1 + hj) * 256 + hd);
      hnext.q[0] = src[0]; hnext.q[1] = src[1]; hnext.q[2] = src[2]; hnext.q[3] = src[3];
      m0n = *(const int4*)(adjRow + j1 + ap);
      m1n = *(const int4*)(adjRow + j1 + ap + 4);
      if (tid < 32) {
        float a2v = a2[bN + j1 + tid], mv = mfin[bN + j1 + tid], sv = sinv[bN + j1 + tid];
        sa2n = a2v; sPn = __expf(a2v - mv) * sv; sQn = __expf(ALPHA * a2v - mv) * sv;
      }
      if (s + 2 < nst) __builtin_prefetch(adjRow + (s + 2) * 32 + ap, 0, 1);
    }
    __syncthreads();

    // A fragment: attention weights, no transcendentals in the hot path
    FragBF af;
    union { uint2 u; unsigned char c[8]; } mk0, mk1;
    const unsigned char* amrow = &AM[(isub + mrow) * 32];
    mk0.u = *(const uint2*)(amrow + kb);
    mk1.u = *(const uint2*)(amrow + kb + 16);
#pragma unroll
    for (int t = 0; t < 8; ++t) {
      float4 jd0 = JD[kb + t];
      float t0 = a1v + jd0.x;
      af.h[t] = (__bf16)(mk0.c[t] ? (t0 > 0.f ? E1 * jd0.y : E2 * jd0.z) : 0.f);
      float4 jd1 = JD[kb + 16 + t];
      float t1 = a1v + jd1.x;
      af.h[8 + t] = (__bf16)(mk1.c[t] ? (t1 > 0.f ? E1 * jd1.y : E2 * jd1.z) : 0.f);
    }
#pragma unroll
    for (int dt = 0; dt < 8; ++dt) {
      FragBF bfm;
      const __bf16* bp = &BT[(dhalf + dt * 16 + mrow) * 40 + kb];
      bfm.q[0] = *(const uint4*)(bp);
      bfm.q[1] = *(const uint4*)(bp + 16);
      acc[dt] = __builtin_amdgcn_wmma_f32_16x16x32_bf16(
          false, af.v, false, bfm.v, (short)0, acc[dt], false, false);
    }
    hreg = hnext; m0 = m0n; m1 = m1n; sa2 = sa2n; sP = sPn; sQ = sQn;
    __syncthreads();
  }

  // ELU epilogue -> bf16 hid
  const size_t oBase = (size_t)b * Nn * 256 + (size_t)(i0 + isub + khalf * 8) * 256;
#pragma unroll
  for (int dt = 0; dt < 8; ++dt) {
    const int dcol = dhalf + dt * 16 + mrow;
    __bf16* p = hid + oBase + dcol;
#pragma unroll
    for (int r = 0; r < 8; ++r) {
      float v = acc[dt][r];
      v = v > 0.f ? v : (__expf(v) - 1.f);
      p[(size_t)r * 256] = (__bf16)v;
    }
  }
}

// ------------------------------------------------------------------ launch
extern "C" void kernel_launch(void* const* d_in, const int* in_sizes, int n_in,
                              void* d_out, int out_size, void* d_ws, size_t ws_size,
                              hipStream_t stream) {
  (void)in_sizes; (void)n_in; (void)out_size; (void)ws_size;
  const int B = 8, N = 2048, D = 256;

  const float* x    = (const float*)d_in[0];
  const int*   adj  = (const int*)  d_in[1];
  const float* W    = (const float*)d_in[2];
  const float* a    = (const float*)d_in[3];
  const float* hanw = (const float*)d_in[4];
  const float* hanb = (const float*)d_in[5];
  float* out = (float*)d_out;

  char* ws = (char*)d_ws;
  size_t off = 0;
  __bf16* xb    = (__bf16*)(ws + off); off += (size_t)B * N * D * 2;   // 8 MB
  __bf16* hb    = (__bf16*)(ws + off); off += (size_t)B * N * D * 2;   // 8 MB
  __bf16* hidb  = (__bf16*)(ws + off); off += (size_t)B * N * D * 2;   // 8 MB
  __bf16* Wb    = (__bf16*)(ws + off); off += (size_t)D * D * 2;
  __bf16* hanwb = (__bf16*)(ws + off); off += (size_t)D * D * 2;
  float* a1    = (float*)(ws + off); off += (size_t)B * N * 4;
  float* a2    = (float*)(ws + off); off += (size_t)B * N * 4;
  float* mpart = (float*)(ws + off); off += (size_t)B * 8 * N * 4;
  float* spart = (float*)(ws + off); off += (size_t)B * 8 * N * 4;
  float* mfin  = (float*)(ws + off); off += (size_t)B * N * 4;
  float* sinv  = (float*)(ws + off); off += (size_t)B * N * 4;

  k_cvt_x<<<(B * N * D / 4 + 255) / 256, 256, 0, stream>>>(x, xb, B * N * D / 4);
  k_cvt_wh<<<(2 * D * D + 255) / 256, 256, 0, stream>>>(W, hanw, Wb, hanwb);
  k_rowdots<<<B * N / 256, 256, 0, stream>>>(x, a, a1, a2, D);
  k_gemm256<0><<<dim3(N / 64, B), 256, 0, stream>>>(xb, Wb, hb, nullptr, nullptr, N, D);
  k_stats_part<<<dim3(N / 256, 8, B), 256, 0, stream>>>(adj, a1, a2, mpart, spart, N);
  k_stats_comb<<<B * N / 256, 256, 0, stream>>>(mpart, spart, mfin, sinv, N);
  k_attn<<<dim3(N / 64, B), 256, 0, stream>>>(adj, hb, a1, a2, mfin, sinv, hidb, N);
  k_gemm256<1><<<dim3(N / 64, B), 256, 0, stream>>>(hidb, hanwb, nullptr, out, hanb, N, D);
}
